// BCMLSTM_60103772340467
// MI455X (gfx1250) — matrix operand order repaired and checked by reading, not compiled, not run on hardware
//
#include <hip/hip_runtime.h>
#include <math.h>

// Problem constants (from reference)
#define T_STEPS 1000
#define B_SZ 32
#define D_SZ 512
#define H_SZ 1024
#define G_SZ 4096 // 4*H
#define DROP_KEEP 0.8f
#define LN_EPS 1e-6f
#define NBLK_SCAN 64

typedef __bf16 bf16_t;
typedef bf16_t v16bf __attribute__((ext_vector_type(16)));
typedef bf16_t v8bf  __attribute__((ext_vector_type(8)));
typedef float  v8f   __attribute__((ext_vector_type(8)));
typedef unsigned int v4u __attribute__((ext_vector_type(4)));
typedef int v8i __attribute__((ext_vector_type(8)));
typedef int v4i __attribute__((ext_vector_type(4)));

union AFrag { v16bf f; v8bf h[2]; };

__device__ __forceinline__ v8f wmma_bf16(v16bf a, v16bf b, v8f c) {
  // D(f32 16x16) = A(bf16 16x32) * B(bf16 32x16) + C
  return __builtin_amdgcn_wmma_f32_16x16x32_bf16(
      /*neg_a=*/false, a, /*neg_b=*/false, b,
      /*c_mod=*/(short)0, c, /*reuse_a=*/false, /*reuse_b=*/false);
}

// ---------------------------------------------------------------------------
// TDM: 2-D tile copy global -> LDS via Tensor Data Mover.
// D# layout per CDNA5 ISA ch.8: group0 {count,lds_addr,global_addr,type=2},
// group1 {data_size=8B, tensor dims, tile dims, dim0 stride}. rows x cols are
// in 8-byte units; rows <= 65535, cols <= 65535.
// This toolchain exposes the 6-arg builtin:
//   (uint32x4 g0, int32x8 g1, int32x4 g2, int32x4 g3, int32x8 aux, i32 cpol)
// ---------------------------------------------------------------------------
__device__ __forceinline__ void tdm_load_2d(unsigned lds_off, const void* gsrc,
                                            unsigned rows, unsigned cols_u,
                                            unsigned stride_u) {
  unsigned long long ga = (unsigned long long)(size_t)gsrc;
  v4u g0;
  g0[0] = 1u;                                   // count=1, no gather
  g0[1] = lds_off;                              // lds_addr (bytes)
  g0[2] = (unsigned)(ga & 0xFFFFFFFFu);         // global_addr[31:0]
  g0[3] = (unsigned)((ga >> 32) & 0x1FFFFFFu)   // global_addr[56:32]
          | (2u << 30);                         // type=2 (image)
  v8i g1;
  g1[0] = (int)(3u << 16);                      // data_size = 3 (8 bytes)
  g1[1] = (int)((cols_u & 0xFFFFu) << 16);      // tensor_dim0[15:0]
  g1[2] = (int)(((cols_u >> 16) & 0xFFFFu) |    // tensor_dim0[31:16]
                ((rows & 0xFFFFu) << 16));      // tensor_dim1[15:0]
  g1[3] = (int)(((rows >> 16) & 0xFFFFu) |      // tensor_dim1[31:16]
                ((cols_u & 0xFFFFu) << 16));    // tile_dim0 = cols
  g1[4] = (int)(rows & 0xFFFFu);                // tile_dim1 = rows, tile_dim2=0
  g1[5] = (int)stride_u;                        // tensor_dim0_stride[31:0]
  g1[6] = 0;                                    // stride hi / dim1_stride lo
  g1[7] = 0;
  v4i z4 = {0, 0, 0, 0};                        // groups 2/3 unused (2-D)
  v8i z8 = {0, 0, 0, 0, 0, 0, 0, 0};
  __builtin_amdgcn_tensor_load_to_lds(g0, g1, z4, z4, z8, 0);
}

// ---------------------------------------------------------------------------
// Device-wide generation barrier (persistent kernel, all blocks resident).
// ---------------------------------------------------------------------------
__device__ __forceinline__ void gsync(unsigned* cnt, unsigned* gen,
                                      unsigned phase, int nblk) {
  __syncthreads();
  if (threadIdx.x == 0) {
    __threadfence();
    unsigned prev = __hip_atomic_fetch_add(cnt, 1u, __ATOMIC_ACQ_REL,
                                           __HIP_MEMORY_SCOPE_AGENT);
    if (prev == (unsigned)nblk * (phase + 1u) - 1u) {
      __hip_atomic_store(gen, phase + 1u, __ATOMIC_RELEASE,
                         __HIP_MEMORY_SCOPE_AGENT);
    } else {
      while (__hip_atomic_load(gen, __ATOMIC_ACQUIRE,
                               __HIP_MEMORY_SCOPE_AGENT) <= phase) {
        __builtin_amdgcn_s_sleep(2);
      }
    }
  }
  __syncthreads();
}

// ---------------------------------------------------------------------------
// Kernel 0: gather block-circulant weights into transposed bf16 matrices
// (N-major so B-operand fragments are contiguous), convert x to bf16.
// ---------------------------------------------------------------------------
__global__ void prep_kernel(const float* __restrict__ x,
                            const float* __restrict__ vih,
                            const float* __restrict__ vhh,
                            const int* __restrict__ idx_ih,
                            const int* __restrict__ idx_hh,
                            bf16_t* __restrict__ WihT,
                            bf16_t* __restrict__ WhhT,
                            bf16_t* __restrict__ xbf) {
  const long NIH = (long)G_SZ * D_SZ;
  const long NHH = (long)G_SZ * H_SZ;
  const long NXB = (long)T_STEPS * B_SZ * D_SZ;
  long i = (long)blockIdx.x * 256 + threadIdx.x;
  if (i < NIH) {
    int g = (int)(i / D_SZ), d = (int)(i % D_SZ);
    WihT[i] = (bf16_t)vih[idx_ih[(long)d * G_SZ + g]];
  } else if (i < NIH + NHH) {
    long j = i - NIH;
    int g = (int)(j / H_SZ), h = (int)(j % H_SZ);
    WhhT[j] = (bf16_t)vhh[idx_hh[(long)h * G_SZ + g]];
  } else if (i < NIH + NHH + NXB) {
    long j = i - NIH - NHH;
    xbf[j] = (bf16_t)x[j];
  }
}

// ---------------------------------------------------------------------------
// Kernel 1: input GEMM  wgates[T*B,4H] = x[T*B,D] @ W_ih[D,4H]  (bf16->f32)
// One wave computes a 16x64 tile: 4 accumulators sharing one A fragment.
// ---------------------------------------------------------------------------
__global__ void igemm_kernel(const bf16_t* __restrict__ xbf,
                             const bf16_t* __restrict__ WihT,
                             float* __restrict__ wgates) {
  int wv = threadIdx.x >> 5;
  int lane = threadIdx.x & 31;
  int lhalf = lane >> 4, lm = lane & 15;
  int task = blockIdx.x * 8 + wv;   // 128000 wave tasks
  int mt = task >> 6;               // 0..1999 (16-row M tile)
  int j64 = task & 63;              // 0..63  (64-col N super-tile)

  const bf16_t* arow = xbf + (size_t)(mt * 16 + lm) * D_SZ;
  v8f acc0 = {}, acc1 = {}, acc2 = {}, acc3 = {};
  for (int kk = 0; kk < D_SZ; kk += 32) {
    AFrag a;
    a.h[0] = *(const v8bf*)(arow + kk + 16 * lhalf);       // K +0..7 / +16..23
    a.h[1] = *(const v8bf*)(arow + kk + 16 * lhalf + 8);   // K +8..15 / +24..31
    const bf16_t* bb =
        WihT + (size_t)(j64 * 64 + lm) * D_SZ + kk + 16 * lhalf;
    v16bf b0 = *(const v16bf*)(bb);
    v16bf b1 = *(const v16bf*)(bb + 16 * (size_t)D_SZ);
    v16bf b2 = *(const v16bf*)(bb + 32 * (size_t)D_SZ);
    v16bf b3 = *(const v16bf*)(bb + 48 * (size_t)D_SZ);
    acc0 = wmma_bf16(a.f, b0, acc0);
    acc1 = wmma_bf16(a.f, b1, acc1);
    acc2 = wmma_bf16(a.f, b2, acc2);
    acc3 = wmma_bf16(a.f, b3, acc3);
  }
  // C layout: VGPR r -> row r + 8*lhalf, lane%16 -> column
  float* orow = wgates + (size_t)(mt * 16) * G_SZ + j64 * 64 + lm;
  for (int r = 0; r < 8; ++r) {
    size_t ro = (size_t)(r + 8 * lhalf) * G_SZ;
    orow[ro + 0]  = acc0[r];
    orow[ro + 16] = acc1[r];
    orow[ro + 32] = acc2[r];
    orow[ro + 48] = acc3[r];
  }
}

// ---------------------------------------------------------------------------
// Kernel 2: persistent recurrent scan. 64 blocks * 8 waves = 512 waves
// == 512 16x16 tiles of u = h @ W_hh per step.
// TDM stages this block's static 64-row W_hh^T slice (128 KB) into LDS once,
// and the live h matrix (64 KB) into LDS every step; inner loop is LDS-only.
// ---------------------------------------------------------------------------
__global__ void scan_kernel(const float* __restrict__ wgates,
                            const bf16_t* __restrict__ WhhT,
                            const float* __restrict__ gamma,
                            const float* __restrict__ beta,
                            bf16_t* __restrict__ hbf,
                            float* __restrict__ cst,
                            float* __restrict__ u,
                            float* __restrict__ hraw,
                            unsigned* __restrict__ bar,
                            float* __restrict__ out) {
  __shared__ bf16_t lds_w[64 * H_SZ];   // 128 KB: W_hh^T rows blk*64..blk*64+63
  __shared__ bf16_t lds_h[B_SZ * H_SZ]; // 64 KB : h (bf16)

  int wv = threadIdx.x >> 5, lane = threadIdx.x & 31;
  int lhalf = lane >> 4, lm = lane & 15;
  int gw = blockIdx.x * 8 + wv;        // 0..511
  int mt = wv & 1;                     // batch half (rows 0-15 / 16-31)
  int ntl = wv >> 1;                   // local col tile 0..3
  int nt = blockIdx.x * 4 + ntl;       // global col tile 0..255
  int tid = blockIdx.x * 256 + threadIdx.x; // 0..16383
  unsigned* cnt = bar;
  unsigned* gen = bar + 64;            // separate cache line
  unsigned phase = 0;

  // Stage this block's W_hh^T slice (64 contiguous rows) once via TDM.
  if (wv == 0) {
    tdm_load_2d((unsigned)(size_t)lds_w,
                WhhT + (size_t)blockIdx.x * 64 * H_SZ,
                /*rows=*/64, /*cols_u=*/H_SZ * 2 / 8, /*stride_u=*/H_SZ * 2 / 8);
    __builtin_amdgcn_s_wait_tensorcnt(0);
  }
  __syncthreads();

  const bf16_t* aL = lds_h + (size_t)(mt * 16 + lm) * H_SZ;
  const bf16_t* bL = lds_w + (size_t)(ntl * 16 + lm) * H_SZ + 16 * lhalf;
  float* up = u + (size_t)(mt * 16) * G_SZ + nt * 16 + lm;

  for (int t = 0; t < T_STEPS; ++t) {
    // ---- 0. TDM: pull current h (bf16) into LDS ----
    if (wv == 0) {
      tdm_load_2d((unsigned)(size_t)lds_h, hbf,
                  /*rows=*/B_SZ, /*cols_u=*/H_SZ * 2 / 8,
                  /*stride_u=*/H_SZ * 2 / 8);
      __builtin_amdgcn_s_wait_tensorcnt(0);
    }
    __syncthreads();

    // ---- 1. u = h @ W_hh for this wave's 16x16 tile (K = 1024, LDS only) ----
    v8f acc = {};
    for (int kk = 0; kk < H_SZ; kk += 32) {
      AFrag a;
      a.h[0] = *(const v8bf*)(aL + kk + 16 * lhalf);
      a.h[1] = *(const v8bf*)(aL + kk + 16 * lhalf + 8);
      v16bf b = *(const v16bf*)(bL + kk);
      acc = wmma_bf16(a.f, b, acc);
    }
    for (int r = 0; r < 8; ++r)
      up[(size_t)(r + 8 * lhalf) * G_SZ] = acc[r];
    gsync(cnt, gen, phase, NBLK_SCAN); ++phase;

    // ---- 2. gates + cell update (2 (b,h) pairs per thread) ----
    const float* wg_t = wgates + (size_t)t * B_SZ * G_SZ;
    for (int j = 0; j < 2; ++j) {
      int p = tid + j * (NBLK_SCAN * 256);
      int b = p >> 10, h = p & 1023;
      const float* ur = u + (size_t)b * G_SZ + h;
      const float* gr = wg_t + (size_t)b * G_SZ + h;
      float gf = gr[0 * H_SZ] + ur[0 * H_SZ];
      float gi = gr[1 * H_SZ] + ur[1 * H_SZ];
      float go = gr[2 * H_SZ] + ur[2 * H_SZ];
      float gc = gr[3 * H_SZ] + ur[3 * H_SZ];
      float ft = 1.f / (1.f + __expf(-gf));
      float it = 1.f / (1.f + __expf(-gi));
      float ot = 1.f / (1.f + __expf(-go));
      float ct = it * tanhf(gc) * DROP_KEEP + ft * cst[p];
      cst[p] = ct;
      hraw[p] = ot * tanhf(ct);
    }
    gsync(cnt, gen, phase, NBLK_SCAN); ++phase;

    // ---- 3. LayerNorm: one wave per batch row (waves 0..31) ----
    if (gw < B_SZ) {
      int b = gw;
      const float* hr = hraw + (size_t)b * H_SZ;
      float s = 0.f, s2 = 0.f;
      for (int j = 0; j < H_SZ / 32; ++j) {
        float v = hr[j * 32 + lane];
        s += v; s2 += v * v;
      }
      for (int m = 16; m >= 1; m >>= 1) {
        s  += __shfl_xor(s,  m, 32);
        s2 += __shfl_xor(s2, m, 32);
      }
      float mean = s * (1.f / H_SZ);
      float var  = s2 * (1.f / H_SZ) - mean * mean;
      float stdv = sqrtf(var * ((float)H_SZ / (float)(H_SZ - 1)));
      float inv  = 1.f / (stdv + LN_EPS);
      float* orow = out + ((size_t)t * B_SZ + b) * H_SZ;
      bf16_t* hnext = hbf + (size_t)b * H_SZ;
      for (int j = 0; j < H_SZ / 32; ++j) {
        int h = j * 32 + lane;
        float v = (hr[h] - mean) * inv * gamma[h] + beta[h];
        orow[h] = v;
        hnext[h] = (bf16_t)v;
      }
    }
    gsync(cnt, gen, phase, NBLK_SCAN); ++phase;
  }
}

// ---------------------------------------------------------------------------
extern "C" void kernel_launch(void* const* d_in, const int* in_sizes, int n_in,
                              void* d_out, int out_size, void* d_ws,
                              size_t ws_size, hipStream_t stream) {
  const float* x     = (const float*)d_in[0];
  const float* vih   = (const float*)d_in[1];
  const float* vhh   = (const float*)d_in[2];
  const float* gamma = (const float*)d_in[3];
  const float* beta  = (const float*)d_in[4];
  const int* idx_ih  = (const int*)d_in[5];
  const int* idx_hh  = (const int*)d_in[6];

  char* ws = (char*)d_ws;
  size_t off = 0;
  auto take = [&](size_t bytes) -> char* {
    char* p = ws + off;
    off = (off + bytes + 255) & ~(size_t)255;
    return p;
  };
  bf16_t* WihT   = (bf16_t*)take((size_t)G_SZ * D_SZ * sizeof(bf16_t));
  bf16_t* WhhT   = (bf16_t*)take((size_t)G_SZ * H_SZ * sizeof(bf16_t));
  bf16_t* xbf    = (bf16_t*)take((size_t)T_STEPS * B_SZ * D_SZ * sizeof(bf16_t));
  float*  wgates = (float*)take((size_t)T_STEPS * B_SZ * G_SZ * sizeof(float));
  float*  u      = (float*)take((size_t)B_SZ * G_SZ * sizeof(float));
  float*  hraw   = (float*)take((size_t)B_SZ * H_SZ * sizeof(float));
  bf16_t* hbf    = (bf16_t*)take((size_t)B_SZ * H_SZ * sizeof(bf16_t));
  float*  cst    = (float*)take((size_t)B_SZ * H_SZ * sizeof(float));
  unsigned* bar  = (unsigned*)take(512);

  // Per-launch state init (graph-capture safe)
  (void)hipMemsetAsync(hbf, 0, (size_t)B_SZ * H_SZ * sizeof(bf16_t), stream);
  (void)hipMemsetAsync(cst, 0, (size_t)B_SZ * H_SZ * sizeof(float), stream);
  (void)hipMemsetAsync(bar, 0, 512, stream);

  long totalPrep = (long)G_SZ * D_SZ + (long)G_SZ * H_SZ +
                   (long)T_STEPS * B_SZ * D_SZ;
  prep_kernel<<<(int)((totalPrep + 255) / 256), 256, 0, stream>>>(
      x, vih, vhh, idx_ih, idx_hh, WihT, WhhT, xbf);

  // 2000 M-tiles * 64 N64-tiles = 128000 waves / 8 waves per block
  igemm_kernel<<<16000, 256, 0, stream>>>(xbf, WihT, wgates);

  scan_kernel<<<NBLK_SCAN, 256, 0, stream>>>(
      wgates, WhhT, gamma, beta, hbf, cst, u, hraw, bar, (float*)d_out);
}
